// Attention_1743756722853
// MI455X (gfx1250) — matrix operand order repaired and checked by reading, not compiled, not run on hardware
//
#include <hip/hip_runtime.h>
#include <hip/hip_bf16.h>

typedef __attribute__((ext_vector_type(16))) _Float16 v16h;
typedef __attribute__((ext_vector_type(8)))  _Float16 v8h;
typedef __attribute__((ext_vector_type(8)))  float    v8f;
typedef __attribute__((ext_vector_type(4)))  float    v4f;

#define BH   64
#define SEQ  2048
#define HD   64
#define KT   64                 // K/V tile width per iteration
#define WAVES 8
#define QW   16                 // query rows per wave
#define QB   (WAVES * QW)       // 128 query rows per block

// convert 8 contiguous f32 -> 8 f16
__device__ __forceinline__ void cvt8(const float* __restrict__ s, _Float16* __restrict__ d) {
    v4f a = *(const v4f*)(s);
    v4f b = *(const v4f*)(s + 4);
#pragma unroll
    for (int i = 0; i < 4; ++i) { d[i] = (_Float16)a[i]; d[i + 4] = (_Float16)b[i]; }
}

__global__ __launch_bounds__(256) void fa_fwd_kernel(
    const float* __restrict__ Q, const float* __restrict__ Kg,
    const float* __restrict__ V, const unsigned char* __restrict__ Msk,
    float* __restrict__ Out)
{
    __shared__ __align__(16) _Float16      sK  [KT * HD];          // K tile, row-major [k][d]
    __shared__ __align__(16) _Float16      sVt [HD * KT];          // V tile, transposed [d][k]
    __shared__ __align__(16) _Float16      sP  [WAVES * QW * KT];  // per-wave P slab [row][k]
    __shared__ __align__(16) unsigned char sM  [QB * KT];          // mask tile [qrow][k]

    const int tid  = threadIdx.x;
    const int wave = tid >> 5;
    const int lane = tid & 31;
    const int m    = lane & 15;   // column / row-in-tile index
    const int hi   = lane >> 4;   // half-wave group
    const int bh   = blockIdx.y;
    const int qrow0 = blockIdx.x * QB;          // block's first query row
    const int qbase = qrow0 + wave * QW;        // wave's first query row

    const float* Qb = Q  + ((size_t)bh * SEQ + qbase) * HD;
    const float* Kb = Kg + (size_t)bh * SEQ * HD;
    const float* Vb = V  + (size_t)bh * SEQ * HD;
    const unsigned char* Mb = Msk + ((size_t)bh * SEQ + qrow0) * SEQ;
    _Float16* sp = &sP[wave * QW * KT];

    // ---- Q A-fragments: 16x64 rows, two k-steps of 32 (ISA 16-bit A layout) ----
    // lane (m,hi): halves 0..7 <- Q[m, 32s + 8hi .. +8), halves 8..15 <- Q[m, 32s + 16 + 8hi .. +8)
    v16h aQ[2];
    {
        const float* qr = Qb + (size_t)m * HD;
#pragma unroll
        for (int s = 0; s < 2; ++s) {
            union { v16h v; _Float16 e[16]; } u;
            const float* p0 = qr + 32 * s + 8 * hi;
            cvt8(p0,      &u.e[0]);
            cvt8(p0 + 16, &u.e[8]);
            aQ[s] = u.v;
        }
    }

    v8f acc[4];
#pragma unroll
    for (int nt = 0; nt < 4; ++nt) acc[nt] = (v8f){};
    float mrun[8], lrun[8];
#pragma unroll
    for (int r = 0; r < 8; ++r) { mrun[r] = -3.0e38f; lrun[r] = 0.0f; }
    const float scale = 0.03125f;   // 1/sqrt(1024)

#pragma unroll 1
    for (int kt = 0; kt < SEQ; kt += KT) {
        __syncthreads();   // previous tile (sK/sVt/sM) fully consumed

        // ---- async DMA: stage 128x64B mask tile into LDS (no conversion needed) ----
        // 512 x 16B chunks; 256 threads x 2 fully-coalesced b128 async copies.
#pragma unroll
        for (int i = 0; i < 2; ++i) {
            const int chunk = i * 256 + tid;
            const int row   = chunk >> 2;
            const int off   = (chunk & 3) << 4;
            unsigned long long ga =
                (unsigned long long)(Mb + (size_t)row * SEQ + (size_t)kt + off);
            unsigned ldsoff = (unsigned)(size_t)(void*)&sM[row * KT + off];
            asm volatile("global_load_async_to_lds_b128 %0, %1, off"
                         :: "v"(ldsoff), "v"(ga) : "memory");
        }

        // ---- cooperative stage of K,V tiles (f32 -> f16), overlaps async DMA ----
        {
            const int r  = tid >> 2;
            const int c0 = (tid & 3) << 4;
            const float* kr = Kb + (size_t)(kt + r) * HD + c0;
            const float* vr = Vb + (size_t)(kt + r) * HD + c0;
            union { v8h v; _Float16 e[8]; } k0, k1, v0, v1;
            cvt8(kr,     k0.e); cvt8(kr + 8, k1.e);
            cvt8(vr,     v0.e); cvt8(vr + 8, v1.e);
            *(v8h*)&sK[r * HD + c0]     = k0.v;
            *(v8h*)&sK[r * HD + c0 + 8] = k1.v;
#pragma unroll
            for (int j = 0; j < 8; ++j) {
                sVt[(c0 + j)     * KT + r] = v0.e[j];
                sVt[(c0 + j + 8) * KT + r] = v1.e[j];
            }
            if (kt + KT < SEQ) {                    // global_prefetch_b8 next tile
                __builtin_prefetch(kr + KT * HD, 0, 1);
                __builtin_prefetch(vr + KT * HD, 0, 1);
            }
        }
        asm volatile("s_wait_asynccnt 0x0" ::: "memory");   // mask DMA landed in LDS
        __syncthreads();

        // ---- scores S = Q . K^T for 4 column tiles, then scale+mask ----
        float p[4][8];
        const unsigned char* smw = &sM[(wave * QW + 8 * hi) * KT + m];
#pragma unroll
        for (int nt = 0; nt < 4; ++nt) {
            // B fragments (ISA 16-bit B layout): col n = nt*16+m, halves 0..7 = K[n][32s+8hi .. +8), 8..15 = K[n][32s+16+8hi .. +8)
            union { v16h v; v8h h[2]; } b0, b1;
            const _Float16* bp = &sK[(nt * 16 + m) * HD + 8 * hi];
            b0.h[0] = *(const v8h*)(bp);
            b0.h[1] = *(const v8h*)(bp + 16);
            b1.h[0] = *(const v8h*)(bp + 32);
            b1.h[1] = *(const v8h*)(bp + 48);
            v8f c = (v8f){};
            c = __builtin_amdgcn_wmma_f32_16x16x32_f16(false, aQ[0], false, b0.v,
                                                       (short)0, c, false, false);
            c = __builtin_amdgcn_wmma_f32_16x16x32_f16(false, aQ[1], false, b1.v,
                                                       (short)0, c, false, false);
#pragma unroll
            for (int r = 0; r < 8; ++r) {
                // C layout: element (row = r + 8*hi, col = nt*16 + m)
                unsigned char mk = smw[r * KT + nt * 16];
                p[nt][r] = mk ? -1.0e9f : c[r] * scale;
            }
        }

        // ---- online softmax: row reductions across the 16-lane group ----
        float rmax[8];
#pragma unroll
        for (int r = 0; r < 8; ++r)
            rmax[r] = fmaxf(fmaxf(p[0][r], p[1][r]), fmaxf(p[2][r], p[3][r]));
#pragma unroll
        for (int off = 1; off < 16; off <<= 1)
#pragma unroll
            for (int r = 0; r < 8; ++r)
                rmax[r] = fmaxf(rmax[r], __shfl_xor(rmax[r], off, 32));

        float rsum[8];
#pragma unroll
        for (int r = 0; r < 8; ++r) {
            float mnew  = fmaxf(mrun[r], rmax[r]);
            float alpha = __expf(mrun[r] - mnew);
            mrun[r] = mnew;
            lrun[r] *= alpha;
#pragma unroll
            for (int nt = 0; nt < 4; ++nt) acc[nt][r] *= alpha;
            float s = 0.0f;
#pragma unroll
            for (int nt = 0; nt < 4; ++nt) {
                float e = __expf(p[nt][r] - mnew);
                p[nt][r] = e;
                s += e;
            }
            rsum[r] = s;
        }
#pragma unroll
        for (int off = 1; off < 16; off <<= 1)
#pragma unroll
            for (int r = 0; r < 8; ++r)
                rsum[r] += __shfl_xor(rsum[r], off, 32);
#pragma unroll
        for (int r = 0; r < 8; ++r) lrun[r] += rsum[r];

        // ---- spill P (f16) to per-wave LDS slab (C layout -> row-major) ----
#pragma unroll
        for (int nt = 0; nt < 4; ++nt)
#pragma unroll
            for (int r = 0; r < 8; ++r)
                sp[(r + 8 * hi) * KT + nt * 16 + m] = (_Float16)p[nt][r];
        // same-wave LDS ops are in-order; compiler inserts s_wait_dscnt before reads

        // ---- O += P (16x64) . V (64x64), contraction over tile k ----
#pragma unroll
        for (int s = 0; s < 2; ++s) {
            union { v16h v; v8h h[2]; } a;   // A layout from LDS
            const _Float16* ap = &sp[m * KT + 32 * s + 8 * hi];
            a.h[0] = *(const v8h*)(ap);
            a.h[1] = *(const v8h*)(ap + 16);
#pragma unroll
            for (int nt = 0; nt < 4; ++nt) {
                union { v16h v; v8h h[2]; } b; // B from transposed V tile
                const _Float16* bp = &sVt[(nt * 16 + m) * KT + 32 * s + 8 * hi];
                b.h[0] = *(const v8h*)(bp);
                b.h[1] = *(const v8h*)(bp + 16);
                acc[nt] = __builtin_amdgcn_wmma_f32_16x16x32_f16(false, a.v, false, b.v,
                                                                 (short)0, acc[nt], false, false);
            }
        }
    }

    // ---- finalize: divide by row sum, store f32 ----
    float* Ob = Out + ((size_t)bh * SEQ + qbase) * HD;
#pragma unroll
    for (int r = 0; r < 8; ++r) {
        float inv = 1.0f / lrun[r];
#pragma unroll
        for (int nt = 0; nt < 4; ++nt)
            Ob[(size_t)(r + 8 * hi) * HD + nt * 16 + m] = acc[nt][r] * inv;
    }
}

extern "C" void kernel_launch(void* const* d_in, const int* in_sizes, int n_in,
                              void* d_out, int out_size, void* d_ws, size_t ws_size,
                              hipStream_t stream) {
    (void)in_sizes; (void)n_in; (void)out_size; (void)d_ws; (void)ws_size;
    const float* Q = (const float*)d_in[0];
    const float* K = (const float*)d_in[1];
    const float* V = (const float*)d_in[2];
    const unsigned char* M = (const unsigned char*)d_in[3];  // jnp.bool_ -> 1 byte/elem
    float* O = (float*)d_out;

    dim3 grid(SEQ / QB, BH);   // (16, 64): one block per (128 q-rows, batch-head)
    fa_fwd_kernel<<<grid, 256, 0, stream>>>(Q, K, V, M, O);
}
